// MHA_66992899883450
// MI455X (gfx1250) — matrix operand order repaired and checked by reading, not compiled, not run on hardware
//
#include <hip/hip_runtime.h>
#include <hip/hip_bf16.h>

// ---- problem constants (match reference) ----
#define BATCH   4
#define SEQ     2048
#define NH      16
#define HD      128
#define DIN     128              // input feature dim
#define DINNER  2048             // NH*HD
#define MTOT    (BATCH*SEQ)      // 8192
#define ATTN_SCALE 0.08838834764831845f  // 1/sqrt(128)
#define NEG_BIG (-3.0e38f)

// LDS tile row strides (f16 elems), padded +8 to spread banks
#define KROW 136                 // 128 + 8
#define VROW 40                  // 32 + 8

typedef __attribute__((ext_vector_type(16))) _Float16 v16h;
typedef __attribute__((ext_vector_type(8)))  _Float16 v8h;
typedef __attribute__((ext_vector_type(8)))  float    v8f;

union AV { v16h v; v8h h8[2]; };

__device__ __forceinline__ v8f wmma16(v16h a, v16h b, v8f c) {
    return __builtin_amdgcn_wmma_f32_16x16x32_f16(false, a, false, b,
                                                  (short)0, c, false, false);
}

// gfx1250 async copy: global -> LDS, 16B per lane, tracked by ASYNCcnt
__device__ __forceinline__ void async_ld_b128(unsigned ldsoff, const void* gptr) {
    asm volatile("global_load_async_to_lds_b128 %0, %1, off"
                 :: "v"(ldsoff), "v"(gptr) : "memory");
}
__device__ __forceinline__ void wait_asynccnt0() {
    asm volatile("s_wait_asynccnt 0x0" ::: "memory");
}

// ---------------- convert f32 -> f16 ----------------
__global__ void cvt_kernel(const float* __restrict__ src,
                           _Float16* __restrict__ dst, int n) {
    int i = blockIdx.x * 256 + threadIdx.x;
    if (i < n) dst[i] = (_Float16)src[i];
}

// -------- transpose + convert: src[K][N] f32 -> dst[N][K] f16 --------
__global__ void cvt_t_kernel(const float* __restrict__ src,
                             _Float16* __restrict__ dst, int K, int N) {
    int i = blockIdx.x * 256 + threadIdx.x;
    if (i >= K * N) return;
    int n = i / K;
    int k = i - n * K;
    dst[(size_t)n * K + k] = (_Float16)src[(size_t)k * N + n];
}

// ---------------- GEMM: C[M,N] = A[M,K] * Bt[N,K]^T + bias ----------------
// block = 256 threads (8 waves, 2x4), wave tile = 32M x 32N, block tile 64M x 128N
// mode 0: f16 row-major out       (Q, K projections)
// mode 1: f16 per-head transposed (V projection -> Vt[b][h][d][s])
// mode 2: f32 row-major out       (final output projection)
__global__ __launch_bounds__(256)
void gemm_kernel(const _Float16* __restrict__ A, const _Float16* __restrict__ Bt,
                 const float* __restrict__ bias, void* __restrict__ outp,
                 int M, int N, int K, int mode) {
    const int lane = threadIdx.x & 31;
    const int wave = threadIdx.x >> 5;
    const int hid  = lane >> 4;       // half-wave id
    const int l16  = lane & 15;
    const int wm = wave & 1;
    const int wn = wave >> 1;
    const int mbase = blockIdx.x * 64 + wm * 32;
    const int nbase = blockIdx.y * 128 + wn * 32;

    v8f acc[2][2] = {};

    const int ksteps = K >> 5;
    for (int ks = 0; ks < ksteps; ++ks) {
        const int k0 = ks * 32;
        AV a[2], b[2];
#pragma unroll
        for (int i = 0; i < 2; ++i) {
            const _Float16* arow = A + (size_t)(mbase + i * 16 + l16) * K + k0 + hid * 8;
            a[i].h8[0] = *(const v8h*)(arow);
            a[i].h8[1] = *(const v8h*)(arow + 16);
        }
#pragma unroll
        for (int j = 0; j < 2; ++j) {
            const _Float16* brow = Bt + (size_t)(nbase + j * 16 + l16) * K + k0 + hid * 8;
            b[j].h8[0] = *(const v8h*)(brow);
            b[j].h8[1] = *(const v8h*)(brow + 16);
        }
#pragma unroll
        for (int i = 0; i < 2; ++i)
#pragma unroll
            for (int j = 0; j < 2; ++j)
                acc[i][j] = wmma16(a[i].v, b[j].v, acc[i][j]);
    }

#pragma unroll
    for (int j = 0; j < 2; ++j) {
        const int n = nbase + j * 16 + l16;
        const float bv = bias[n];
#pragma unroll
        for (int i = 0; i < 2; ++i) {
#pragma unroll
            for (int r = 0; r < 8; ++r) {
                const int m = mbase + i * 16 + hid * 8 + r;
                const float v = acc[i][j][r] + bv;
                if (mode == 0) {
                    ((_Float16*)outp)[(size_t)m * N + n] = (_Float16)v;
                } else if (mode == 1) {
                    const int bb = m / SEQ, s = m - bb * SEQ;
                    const int hh = n / HD,  d = n - hh * HD;
                    ((_Float16*)outp)[((size_t)(bb * NH + hh) * HD + d) * SEQ + s] = (_Float16)v;
                } else {
                    ((float*)outp)[(size_t)m * N + n] = v;
                }
            }
        }
    }
}

// ---- cooperative async staging of one 32-key K tile and V tile into LDS ----
__device__ __forceinline__ void stage_kv(const _Float16* __restrict__ Kp,
                                         const _Float16* __restrict__ Vt,
                                         _Float16 (*kb)[KROW], _Float16 (*vb)[VROW],
                                         int bb, int h, int kbase, int tid) {
    // K tile: 32 keys x 128 d, 512 x 16B chunks, 4 per thread
#pragma unroll
    for (int j = 0; j < 4; ++j) {
        const int c = tid + j * 128;
        const int row = c >> 4, c16 = c & 15;
        const _Float16* g =
            Kp + ((size_t)bb * SEQ + kbase + row) * DINNER + h * HD + c16 * 8;
        async_ld_b128((unsigned)(size_t)&kb[row][c16 * 8], g);
    }
    // V tile (pre-transposed Vt[b][h][d][s]): 128 d x 32 s, 512 x 16B chunks
#pragma unroll
    for (int j = 0; j < 4; ++j) {
        const int c = tid + j * 128;
        const int d = c >> 2, c4 = c & 3;
        const _Float16* g =
            Vt + ((size_t)(bb * NH + h) * HD + d) * SEQ + kbase + c4 * 8;
        async_ld_b128((unsigned)(size_t)&vb[d][c4 * 8], g);
    }
}

// ---------------- flash attention (causal) ----------------
// grid = (SEQ/64, BATCH*NH), block = 128 (4 waves); each wave owns 16 q-rows.
// K/V tiles staged once per block via async-to-LDS, double-buffered.
__global__ __launch_bounds__(128)
void attn_kernel(const _Float16* __restrict__ Qp, const _Float16* __restrict__ Kp,
                 const _Float16* __restrict__ Vt, _Float16* __restrict__ ctx) {
    __shared__ __align__(16) _Float16 kbuf[2][32][KROW];
    __shared__ __align__(16) _Float16 vbuf[2][128][VROW];
    __shared__ __align__(16) _Float16 pbuf[4][16 * 32];

    const int tid  = threadIdx.x;
    const int lane = tid & 31;
    const int wave = tid >> 5;
    const int hid  = lane >> 4;
    const int l16  = lane & 15;
    const int bh = blockIdx.y;
    const int bb = bh >> 4;
    const int h  = bh & 15;
    const int qbase = blockIdx.x * 64 + wave * 16;

    // load the wave's 16x128 Q tile as four A-operands (K-steps of 32)
    AV qa[4];
    {
        const _Float16* qrow =
            Qp + ((size_t)bb * SEQ + qbase + l16) * DINNER + h * HD;
#pragma unroll
        for (int kd = 0; kd < 4; ++kd) {
            qa[kd].h8[0] = *(const v8h*)(qrow + kd * 32 + hid * 8);
            qa[kd].h8[1] = *(const v8h*)(qrow + kd * 32 + 16 + hid * 8);
        }
    }

    v8f acc[8] = {};                 // 16 rows x 128 d, C-layout per 16-col tile
    float mrow[8], lrow[8];
#pragma unroll
    for (int r = 0; r < 8; ++r) { mrow[r] = NEG_BIG; lrow[r] = 0.0f; }

    const int nkw   = (qbase + 47) >> 5;            // this wave's causal step count
    const int nkblk = (blockIdx.x * 64 + 95) >> 5;  // block max (wave 3)

    // prologue: stage tile 0 into buffer 0
    stage_kv(Kp, Vt, kbuf[0], vbuf[0], bb, h, 0, tid);
    wait_asynccnt0();
    __syncthreads();

    for (int kk = 0; kk < nkblk; ++kk) {
        const int cur = kk & 1;
        if (kk + 1 < nkblk)          // overlap next tile's DMA with compute
            stage_kv(Kp, Vt, kbuf[cur ^ 1], vbuf[cur ^ 1], bb, h, (kk + 1) * 32, tid);

        if (kk < nkw) {
            const int kbase = kk * 32;
            v8f s0 = {}, s1 = {};
            const _Float16* kr0 = &kbuf[cur][l16][hid * 8];
            const _Float16* kr1 = &kbuf[cur][16 + l16][hid * 8];
#pragma unroll
            for (int kd = 0; kd < 4; ++kd) {
                AV b0, b1;
                b0.h8[0] = *(const v8h*)(kr0 + kd * 32);
                b0.h8[1] = *(const v8h*)(kr0 + kd * 32 + 16);
                b1.h8[0] = *(const v8h*)(kr1 + kd * 32);
                b1.h8[1] = *(const v8h*)(kr1 + kd * 32 + 16);
                s0 = wmma16(qa[kd].v, b0.v, s0);
                s1 = wmma16(qa[kd].v, b1.v, s1);
            }

            // online softmax; each row lives across the 16 lanes of one half-wave
#pragma unroll
            for (int r = 0; r < 8; ++r) {
                const int q = qbase + hid * 8 + r;
                float x0 = (kbase + l16      <= q) ? s0[r] * ATTN_SCALE : NEG_BIG;
                float x1 = (kbase + 16 + l16 <= q) ? s1[r] * ATTN_SCALE : NEG_BIG;
                float tmax = fmaxf(x0, x1);
                tmax = fmaxf(tmax, __shfl_xor(tmax, 1, 32));
                tmax = fmaxf(tmax, __shfl_xor(tmax, 2, 32));
                tmax = fmaxf(tmax, __shfl_xor(tmax, 4, 32));
                tmax = fmaxf(tmax, __shfl_xor(tmax, 8, 32));
                const float mnew = fmaxf(mrow[r], tmax);
                const float corr = __expf(mrow[r] - mnew);
                const float p0 = __expf(x0 - mnew);
                const float p1 = __expf(x1 - mnew);
                float ps = p0 + p1;
                ps += __shfl_xor(ps, 1, 32);
                ps += __shfl_xor(ps, 2, 32);
                ps += __shfl_xor(ps, 4, 32);
                ps += __shfl_xor(ps, 8, 32);
                lrow[r] = lrow[r] * corr + ps;
                mrow[r] = mnew;
#pragma unroll
                for (int t = 0; t < 8; ++t) acc[t][r] *= corr;
                _Float16* pb = &pbuf[wave][(hid * 8 + r) * 32];
                pb[l16]      = (_Float16)p0;
                pb[16 + l16] = (_Float16)p1;
            }
            __builtin_amdgcn_wave_barrier();

            // reload P as a 16x32 A-operand
            AV pa;
            {
                const _Float16* pr = &pbuf[wave][l16 * 32 + hid * 8];
                pa.h8[0] = *(const v8h*)(pr);
                pa.h8[1] = *(const v8h*)(pr + 16);
            }

            // ctx += P * V from the LDS V tile
#pragma unroll
            for (int t = 0; t < 8; ++t) {
                AV vb;
                const _Float16* vr = &vbuf[cur][t * 16 + l16][hid * 8];
                vb.h8[0] = *(const v8h*)(vr);
                vb.h8[1] = *(const v8h*)(vr + 16);
                acc[t] = wmma16(pa.v, vb.v, acc[t]);
            }
            __builtin_amdgcn_wave_barrier();
        }

        wait_asynccnt0();   // our share of next tile's DMA done
        __syncthreads();    // everyone done reading cur + next buffer complete
    }

    // normalize and write ctx (f16 row-major [m][h*HD+d])
#pragma unroll
    for (int r = 0; r < 8; ++r) {
        const float inv = 1.0f / lrow[r];
        const int s = qbase + hid * 8 + r;
        _Float16* crow = ctx + ((size_t)bb * SEQ + s) * DINNER + h * HD + l16;
#pragma unroll
        for (int t = 0; t < 8; ++t)
            crow[t * 16] = (_Float16)(acc[t][r] * inv);
    }
}

// ---------------- host-side orchestration ----------------
extern "C" void kernel_launch(void* const* d_in, const int* in_sizes, int n_in,
                              void* d_out, int out_size, void* d_ws, size_t ws_size,
                              hipStream_t stream) {
    const float* query = (const float*)d_in[0];
    const float* Wq = (const float*)d_in[1];
    const float* bq = (const float*)d_in[2];
    const float* Wk = (const float*)d_in[3];
    const float* bk = (const float*)d_in[4];
    const float* Wv = (const float*)d_in[5];
    const float* bv = (const float*)d_in[6];
    const float* Wo = (const float*)d_in[7];
    const float* bo = (const float*)d_in[8];

    // workspace carve (all _Float16)
    _Float16* w    = (_Float16*)d_ws;
    _Float16* Qin  = w;                                 // 8192*128
    _Float16* Wqt  = Qin + (size_t)MTOT * DIN;          // 2048*128
    _Float16* Wkt  = Wqt + (size_t)DINNER * DIN;
    _Float16* Wvt  = Wkt + (size_t)DINNER * DIN;
    _Float16* Wot  = Wvt + (size_t)DINNER * DIN;        // 128*2048
    _Float16* Qp   = Wot + (size_t)HD * DINNER;         // 8192*2048
    _Float16* Kp   = Qp  + (size_t)MTOT * DINNER;
    _Float16* Vtr  = Kp  + (size_t)MTOT * DINNER;
    _Float16* ctxb = Vtr + (size_t)MTOT * DINNER;

    // 1) convert inputs / transpose weights to f16
    {
        int n = MTOT * DIN;
        cvt_kernel<<<(n + 255) / 256, 256, 0, stream>>>(query, Qin, n);
        int nw = DIN * DINNER;
        cvt_t_kernel<<<(nw + 255) / 256, 256, 0, stream>>>(Wq, Wqt, DIN, DINNER);
        cvt_t_kernel<<<(nw + 255) / 256, 256, 0, stream>>>(Wk, Wkt, DIN, DINNER);
        cvt_t_kernel<<<(nw + 255) / 256, 256, 0, stream>>>(Wv, Wvt, DIN, DINNER);
        cvt_t_kernel<<<(nw + 255) / 256, 256, 0, stream>>>(Wo, Wot, DINNER, HD);
    }

    // 2) QKV projections (M=8192, N=2048, K=128)
    {
        dim3 grid(MTOT / 64, DINNER / 128);
        gemm_kernel<<<grid, 256, 0, stream>>>(Qin, Wqt, bq, Qp,  MTOT, DINNER, DIN, 0);
        gemm_kernel<<<grid, 256, 0, stream>>>(Qin, Wkt, bk, Kp,  MTOT, DINNER, DIN, 0);
        gemm_kernel<<<grid, 256, 0, stream>>>(Qin, Wvt, bv, Vtr, MTOT, DINNER, DIN, 1);
    }

    // 3) causal flash attention (async-LDS staged K/V)
    {
        dim3 grid(SEQ / 64, BATCH * NH);
        attn_kernel<<<grid, 128, 0, stream>>>(Qp, Kp, Vtr, ctxb);
    }

    // 4) output projection (M=8192, N=128, K=2048) -> f32 d_out
    {
        dim3 grid(MTOT / 64, HD / 128);
        gemm_kernel<<<grid, 256, 0, stream>>>(ctxb, Wot, bo, d_out, MTOT, HD, DINNER, 2);
    }
}
